// EventModelV2_56642028699917
// MI455X (gfx1250) — matrix operand order repaired and checked by reading, not compiled, not run on hardware
//
#include <hip/hip_runtime.h>
#include <hip/hip_bf16.h>

// ---------------------------------------------------------------------------
// EventModelV2 on gfx1250 (MI455X, wave32, WMMA).
//   k_embed        : token -> f16 embedding (65536x128) + mask output
//   k_prep_weights : Wf|Wb -> f16 (128x512), bf|bb -> f32(512), Uf/Ub -> f16
//   k_prep_expert  : per-batch expert gather W_arg[idx] -> f16 (160x512), b_arg
//   k_fill_ev      : broadcast event embedding into feat[:, :, 128:160] (f16)
//   k_xw_gemm      : Z = emb16 @ [Wf|Wb] + [bf|bb]   (WMMA, M=65536,K=128,N=512)
//   k_lstm         : sequential masked LSTM, 1 block/direction, h@U via WMMA,
//                    software-pipelined double-buffered Z staging
//   k_event_logits : sigmoid(bi[:, -1, :] @ Wt + bt)
//   k_arg_gemm     : out[b,c,s,a] = feat @ WgAll + bg (WMMA, M=65536,K=160,N=512)
// LDS notes: B operands stored transposed ([n][k]) with padded leading dims so
// WMMA fragment loads are contiguous 32B (2x ds_load_b128) hitting 16 distinct
// banks. GEMM epilogues stage the 64x64 f32 block in LDS (reusing lsA) and
// emit fully-coalesced float4 global stores.
// ---------------------------------------------------------------------------

typedef __attribute__((ext_vector_type(16))) _Float16 v16h;
typedef __attribute__((ext_vector_type(8)))  float    v8f;

#define BDIM        64
#define SDIM        1024
#define CEMB        128
#define HID         64
#define GATES       256     // 4*HID
#define NZ          512     // fwd+bwd gate columns
#define EV_EMBED    32
#define ARG_DIM     8
#define FEATD       160     // 2*HID + EV_EMBED
#define NBA         512     // BDIM * ARG_DIM

#define LDA_XW      136     // padded CEMB
#define LDK_ARG     168     // padded FEATD
#define LDH         72      // padded HID (lsH rows / lsUt rows)
#define LDO         68      // padded 64 (f32 epilogue staging)

// ----------------------- WMMA fragment loaders (wave32) ---------------------
// A (16x32 f16): lanes 0-15 hold row M=lane, K in {0..15};
//                lanes 16-31 hold row M=lane-16, K in {16..31}.
__device__ inline v16h load_a_frag(const _Float16* A, int lda, int m_base, int k_base) {
  const int lane = threadIdx.x & 31;
  const int m = m_base + (lane & 15);
  const int hi = lane >> 4;
  const _Float16* p0 = A + m * lda + k_base + (hi ? 16 : 0);
  const _Float16* p1 = A + m * lda + k_base + (hi ? 24 : 8);
  v16h a;
#pragma unroll
  for (int i = 0; i < 8; ++i) { a[i] = p0[i]; a[8 + i] = p1[i]; }
  return a;
}

// B (32x16 f16) from TRANSPOSED LDS storage Bt[n][k] (ldb = padded K).
__device__ inline v16h load_bt_frag(const _Float16* Bt, int ldb, int k_base, int n_base) {
  const int lane = threadIdx.x & 31;
  const int n = n_base + (lane & 15);
  const _Float16* p = Bt + n * ldb + k_base + ((lane >> 4) ? 16 : 0);
  v16h b;
#pragma unroll
  for (int i = 0; i < 16; ++i) b[i] = p[i];
  return b;
}

// ----------------------------- kernels --------------------------------------

__global__ void k_embed(const int* __restrict__ tokens,
                        const float* __restrict__ table,
                        _Float16* __restrict__ emb16,
                        float* __restrict__ mask_out) {
  const int tid = blockIdx.x * blockDim.x + threadIdx.x;   // B*S*128 threads
  const int cs = tid >> 7;
  const int d  = tid & 127;
  const int tok = tokens[cs];
  emb16[tid] = (_Float16)table[tok * CEMB + d];
  if (d == 0) mask_out[cs] = (tok != 0) ? 1.0f : 0.0f;
}

__global__ void k_prep_weights(const float* __restrict__ Wf, const float* __restrict__ bf,
                               const float* __restrict__ Wb, const float* __restrict__ bb,
                               const float* __restrict__ Uf, const float* __restrict__ Ub,
                               _Float16* __restrict__ Wc16, float* __restrict__ biasC,
                               _Float16* __restrict__ U16f, _Float16* __restrict__ U16b) {
  for (int i = blockIdx.x * blockDim.x + threadIdx.x; i < CEMB * NZ;
       i += gridDim.x * blockDim.x) {
    const int k = i >> 9, n = i & 511;
    const float v = (n < GATES) ? Wf[k * GATES + n] : Wb[k * GATES + (n - GATES)];
    Wc16[i] = (_Float16)v;
    if (k == 0) biasC[n] = (n < GATES) ? bf[n] : bb[n - GATES];
    if (i < HID * GATES) { U16f[i] = (_Float16)Uf[i]; U16b[i] = (_Float16)Ub[i]; }
  }
}

__global__ void k_prep_expert(const int* __restrict__ ev_type,
                              const float* __restrict__ W_arg,
                              const float* __restrict__ b_arg,
                              const float* __restrict__ ev_table,
                              _Float16* __restrict__ WgAll, float* __restrict__ bgAll,
                              _Float16* __restrict__ ev16) {
  for (int i = blockIdx.x * blockDim.x + threadIdx.x; i < FEATD * NBA;
       i += gridDim.x * blockDim.x) {
    const int f = i >> 9, n = i & 511;
    const int b = n >> 3, a = n & 7;
    const int id = ev_type[b];
    WgAll[i] = (_Float16)W_arg[(id * FEATD + f) * ARG_DIM + a];
    if (f == 0) bgAll[n] = b_arg[id * ARG_DIM + a];
    if (i < BDIM * EV_EMBED) {
      const int c = i >> 5, j = i & 31;
      ev16[i] = (_Float16)ev_table[ev_type[c] * EV_EMBED + j];
    }
  }
}

__global__ void k_fill_ev(const _Float16* __restrict__ ev16,
                          _Float16* __restrict__ feat16) {
  const int tid = blockIdx.x * blockDim.x + threadIdx.x;   // B*S*32 threads
  const int cs = tid >> 5, j = tid & 31;
  const int c = cs >> 10;
  feat16[(size_t)cs * FEATD + 2 * HID + j] = ev16[c * EV_EMBED + j];
}

// Z = emb16 @ [Wf|Wb] + [bf|bb] ; M=65536, K=128, N=512; 64x64 tile per block.
__global__ __launch_bounds__(256)
void k_xw_gemm(const _Float16* __restrict__ A,
               const _Float16* __restrict__ Bw,
               const float* __restrict__ bias,
               float* __restrict__ Z) {
  __shared__ __attribute__((aligned(16))) _Float16 lsA[64 * LDA_XW];   // 17408 B
  __shared__ __attribute__((aligned(16))) _Float16 lsBt[64 * LDA_XW];  // [n][k]
  float* ls_out = (float*)lsA;    // 64*LDO f32 == 17408 B, reused after compute
  const int mBase = blockIdx.x * 64;
  const int nBase = blockIdx.y * 64;
  {
    const uint4* gA = (const uint4*)(A + (size_t)mBase * CEMB);
    for (int i = threadIdx.x; i < 64 * (CEMB / 8); i += blockDim.x) {
      const int r = i >> 4, q = i & 15;            // CEMB/8 == 16
      *(uint4*)(lsA + r * LDA_XW + q * 8) = gA[i];
    }
  }
  for (int i = threadIdx.x; i < CEMB * 64; i += blockDim.x) {
    const int k = i >> 6, n = i & 63;
    lsBt[n * LDA_XW + k] = Bw[(size_t)k * NZ + nBase + n];
  }
  __syncthreads();
  const int wave = threadIdx.x >> 5;
  const int lane = threadIdx.x & 31;
  v8f acc[2];
#pragma unroll
  for (int t = 0; t < 2; ++t) {
    const int tile = wave * 2 + t;        // 16 tiles in a 4x4 grid
    const int tm = (tile >> 2) * 16;
    const int tn = (tile & 3) * 16;
    acc[t] = (v8f){};
#pragma unroll
    for (int kc = 0; kc < 4; ++kc) {
      v16h a = load_a_frag(lsA, LDA_XW, tm, kc * 32);
      v16h b = load_bt_frag(lsBt, LDA_XW, kc * 32, tn);
      acc[t] = __builtin_amdgcn_wmma_f32_16x16x32_f16(false, a, false, b,
                                                      (short)0, acc[t], false, false);
    }
  }
  __syncthreads();   // done reading lsA/lsBt; reuse lsA as f32 staging
#pragma unroll
  for (int t = 0; t < 2; ++t) {
    const int tile = wave * 2 + t;
    const int tm = (tile >> 2) * 16;
    const int tn = (tile & 3) * 16;
    const int n = tn + (lane & 15);
#pragma unroll
    for (int r = 0; r < 8; ++r) {
      const int m = tm + ((lane < 16) ? r : (r + 8));
      ls_out[m * LDO + n] = acc[t][r];
    }
  }
  __syncthreads();
  // coalesced writeback: 256B contiguous per row
  for (int i = threadIdx.x; i < 64 * 16; i += blockDim.x) {
    const int r = i >> 4, q = i & 15;
    const float4 bv = *(const float4*)(bias + nBase + q * 4);
    float4 v = *(const float4*)(ls_out + r * LDO + q * 4);
    v.x += bv.x; v.y += bv.y; v.z += bv.z; v.w += bv.w;
    *(float4*)(Z + (size_t)(mBase + r) * NZ + nBase + q * 4) = v;
  }
}

// Sequential masked LSTM. blockIdx.x = direction (0 fwd, 1 bwd).
// 1024 threads = 32 waves; 64 output tiles (4 M x 16 N) -> 2 tiles/wave.
// Double-buffered Z staging: global loads for step t+1 issued before the WMMA
// phase of step t and committed to LDS after the gate phase (latency hidden).
__global__ __launch_bounds__(1024)
void k_lstm(const float* __restrict__ Z,
            const _Float16* __restrict__ U16f,
            const _Float16* __restrict__ U16b,
            const int* __restrict__ tokens,
            _Float16* __restrict__ feat16) {
  extern __shared__ char smem[];
  _Float16* lsUt = (_Float16*)smem;                            // 256*72 f16 = 36864B
  _Float16* lsH  = (_Float16*)(smem + 36864);                  //  64*72 f16 =  9216B
  float*    lsC  = (float*)(smem + 36864 + 9216);              //  64*64 f32 = 16384B
  float*    zbuf = (float*)(smem + 36864 + 9216 + 16384);      // 2x 64*256 f32 = 131072B
  int*      lsM  = (int*)(smem + 36864 + 9216 + 16384 + 131072); // 2x 64 i32 = 512B
  const int dir = blockIdx.x;
  const _Float16* U = dir ? U16b : U16f;
  for (int i = threadIdx.x; i < HID * GATES; i += blockDim.x) {
    const int k = i >> 8, n = i & 255;
    lsUt[n * LDH + k] = U[i];                                  // transposed U
  }
  for (int i = threadIdx.x; i < BDIM * LDH; i += blockDim.x) lsH[i] = (_Float16)0.f;
  for (int i = threadIdx.x; i < BDIM * HID; i += blockDim.x) lsC[i] = 0.f;

  const int wave = threadIdx.x >> 5;
  const int lane = threadIdx.x & 31;
  float4 pf[4];
  int pftok = 0;

  auto issue_prefetch = [&](int ts_n) {
#pragma unroll
    for (int j = 0; j < 4; ++j) {
      const int i = threadIdx.x + j * 1024;      // 0..4095
      const int b = i >> 6, q = i & 63;
      pf[j] = *(const float4*)(Z + (size_t)(b * SDIM + ts_n) * NZ + dir * GATES + q * 4);
    }
    if (threadIdx.x < BDIM) pftok = tokens[threadIdx.x * SDIM + ts_n];
  };
  auto commit_prefetch = [&](float* zb, int* mb) {
#pragma unroll
    for (int j = 0; j < 4; ++j) {
      const int i = threadIdx.x + j * 1024;
      const int b = i >> 6, q = i & 63;
      *(float4*)(zb + b * GATES + q * 4) = pf[j];
    }
    if (threadIdx.x < BDIM) mb[threadIdx.x] = pftok;
  };

  // prologue: stage step 0
  issue_prefetch(dir ? (SDIM - 1) : 0);
  commit_prefetch(zbuf, lsM);
  __syncthreads();

  for (int t = 0; t < SDIM; ++t) {
    const int ts = dir ? (SDIM - 1 - t) : t;
    const int cur = t & 1;
    float* zb = zbuf + cur * (BDIM * GATES);
    int*   mb = lsM + cur * BDIM;
    // issue global loads for step t+1 (completion awaited after gate phase)
    if (t + 1 < SDIM) issue_prefetch(dir ? (SDIM - 2 - t) : (t + 1));
    // z += h @ U  (WMMA, C operand chained from zbuf)
#pragma unroll
    for (int tt = 0; tt < 2; ++tt) {
      const int tile = wave * 2 + tt;     // 0..63
      const int tm = (tile & 3) * 16;
      const int tn = (tile >> 2) * 16;
      const int n = tn + (lane & 15);
      v8f c;
#pragma unroll
      for (int r = 0; r < 8; ++r) {
        const int m = tm + ((lane < 16) ? r : (r + 8));
        c[r] = zb[m * GATES + n];
      }
      v16h a0 = load_a_frag(lsH, LDH, tm, 0);
      v16h b0 = load_bt_frag(lsUt, LDH, 0, tn);
      c = __builtin_amdgcn_wmma_f32_16x16x32_f16(false, a0, false, b0,
                                                 (short)0, c, false, false);
      v16h a1 = load_a_frag(lsH, LDH, tm, 32);
      v16h b1 = load_bt_frag(lsUt, LDH, 32, tn);
      c = __builtin_amdgcn_wmma_f32_16x16x32_f16(false, a1, false, b1,
                                                 (short)0, c, false, false);
#pragma unroll
      for (int r = 0; r < 8; ++r) {
        const int m = tm + ((lane < 16) ? r : (r + 8));
        zb[m * GATES + n] = c[r];
      }
    }
    __syncthreads();
    // gates + masked state update; write h into feat
#pragma unroll
    for (int j = 0; j < 4; ++j) {
      const int p = threadIdx.x + j * 1024;    // 0..4095 -> (batch, hidden)
      const int b = p >> 6, h = p & 63;
      const float zi = zb[b * GATES + h];
      const float zf = zb[b * GATES + HID + h];
      const float zg = zb[b * GATES + 2 * HID + h];
      const float zo = zb[b * GATES + 3 * HID + h];
      const float c_old = lsC[p];
      const float h_old = (float)lsH[b * LDH + h];
      const float ig = 1.f / (1.f + __expf(-zi));
      const float fg = 1.f / (1.f + __expf(-zf));
      const float og = 1.f / (1.f + __expf(-zo));
      const float c_new = fg * c_old + ig * tanhf(zg);
      const float h_new = og * tanhf(c_new);
      const bool msk = mb[b] != 0;
      const float c2 = msk ? c_new : c_old;
      const float h2 = msk ? h_new : h_old;
      lsC[p] = c2;
      lsH[b * LDH + h] = (_Float16)h2;
      feat16[(size_t)(b * SDIM + ts) * FEATD + dir * HID + h] = (_Float16)h2;
    }
    // commit prefetched step t+1 into the other buffer
    if (t + 1 < SDIM)
      commit_prefetch(zbuf + (cur ^ 1) * (BDIM * GATES), lsM + (cur ^ 1) * BDIM);
    __syncthreads();
  }
}

__global__ void k_event_logits(const _Float16* __restrict__ feat16,
                               const float* __restrict__ Wt,
                               const float* __restrict__ bt,
                               float* __restrict__ out) {
  const int tid = blockIdx.x * blockDim.x + threadIdx.x;   // 1024 threads
  const int c = tid >> 4, e = tid & 15;
  const _Float16* row = feat16 + (size_t)(c * SDIM + (SDIM - 1)) * FEATD;
  float acc = bt[e];
  for (int f = 0; f < 2 * HID; ++f) acc += (float)row[f] * Wt[f * 16 + e];
  out[tid] = 1.f / (1.f + __expf(-acc));
}

// out[b,c,s,a] = feat[(c,s),:] @ WgAll[:, (b,a)] + bg ; M=65536, K=160, N=512.
__global__ __launch_bounds__(256)
void k_arg_gemm(const _Float16* __restrict__ feat16,
                const _Float16* __restrict__ Wg,
                const float* __restrict__ bg,
                float* __restrict__ out) {
  __shared__ __attribute__((aligned(16))) _Float16 lsA[64 * LDK_ARG];   // 21504 B
  __shared__ __attribute__((aligned(16))) _Float16 lsBt[64 * LDK_ARG];  // [n][k]
  float* ls_out = (float*)lsA;    // 64*LDO f32 = 17408 B, reused after compute
  const int mBase = blockIdx.x * 64;
  const int nBase = blockIdx.y * 64;
  for (int i = threadIdx.x; i < 64 * (FEATD / 8); i += blockDim.x) {   // 64*20
    const int r = i / 20, q = i % 20;
    *(uint4*)(lsA + r * LDK_ARG + q * 8) =
        *(const uint4*)(feat16 + (size_t)(mBase + r) * FEATD + q * 8);
  }
  for (int i = threadIdx.x; i < FEATD * 64; i += blockDim.x) {
    const int k = i >> 6, n = i & 63;
    lsBt[n * LDK_ARG + k] = Wg[(size_t)k * NBA + nBase + n];
  }
  __syncthreads();
  const int wave = threadIdx.x >> 5;
  const int lane = threadIdx.x & 31;
  v8f acc[2];
#pragma unroll
  for (int t = 0; t < 2; ++t) {
    const int tile = wave * 2 + t;
    const int tm = (tile >> 2) * 16;
    const int tn = (tile & 3) * 16;
    acc[t] = (v8f){};
#pragma unroll
    for (int kc = 0; kc < 5; ++kc) {
      v16h a = load_a_frag(lsA, LDK_ARG, tm, kc * 32);
      v16h b = load_bt_frag(lsBt, LDK_ARG, kc * 32, tn);
      acc[t] = __builtin_amdgcn_wmma_f32_16x16x32_f16(false, a, false, b,
                                                      (short)0, acc[t], false, false);
    }
  }
  __syncthreads();   // done reading lsA/lsBt; reuse lsA as f32 staging
#pragma unroll
  for (int t = 0; t < 2; ++t) {
    const int tile = wave * 2 + t;
    const int tm = (tile >> 2) * 16;
    const int tn = (tile & 3) * 16;
    const int n = tn + (lane & 15);
#pragma unroll
    for (int r = 0; r < 8; ++r) {
      const int m = tm + ((lane < 16) ? r : (r + 8));
      ls_out[m * LDO + n] = acc[t][r];
    }
  }
  __syncthreads();
  // coalesced writeback: out region per local bb is 512 contiguous floats:
  // out[((bb*64 + c)*1024 + sBase + sl)*8 + aa], block covers 8 bb values.
  const int c = mBase >> 10;
  const int sBase = mBase & 1023;
  for (int i = threadIdx.x; i < 1024; i += blockDim.x) {   // float4 granules
    const int bbl = i >> 7;            // 0..7
    const int rem = i & 127;           // (sl*8 + aa) / 4
    const int sl = rem >> 1, a4 = (rem & 1) * 4;
    const int n0 = nBase + bbl * 8 + a4;
    const int bb = n0 >> 3;
    float4 v = *(const float4*)(ls_out + sl * LDO + bbl * 8 + a4);
    const float4 bv = *(const float4*)(bg + n0);
    v.x += bv.x; v.y += bv.y; v.z += bv.z; v.w += bv.w;
    *(float4*)(out + (((size_t)bb * BDIM + c) * SDIM + sBase + sl) * ARG_DIM + (n0 & 7)) = v;
  }
}

// ----------------------------- launcher -------------------------------------

extern "C" void kernel_launch(void* const* d_in, const int* in_sizes, int n_in,
                              void* d_out, int out_size, void* d_ws, size_t ws_size,
                              hipStream_t stream) {
  (void)in_sizes; (void)n_in; (void)out_size; (void)ws_size;

  const int*   tokens   = (const int*)d_in[0];
  const int*   ev_type  = (const int*)d_in[1];
  const float* table    = (const float*)d_in[2];
  const float* Wf       = (const float*)d_in[3];
  const float* Uf       = (const float*)d_in[4];
  const float* bf       = (const float*)d_in[5];
  const float* Wb       = (const float*)d_in[6];
  const float* Ub       = (const float*)d_in[7];
  const float* bb       = (const float*)d_in[8];
  const float* ev_table = (const float*)d_in[9];
  const float* Wt       = (const float*)d_in[10];
  const float* bt       = (const float*)d_in[11];
  const float* W_arg    = (const float*)d_in[12];
  const float* b_arg    = (const float*)d_in[13];

  float* out_logits = (float*)d_out;                                  // 64*16
  float* out_args   = out_logits + BDIM * 16;                         // 64*64*1024*8
  float* out_mask   = out_args + (size_t)BDIM * BDIM * SDIM * ARG_DIM;// 64*1024

  // workspace carve-up
  char* ws = (char*)d_ws;
  _Float16* emb16  = (_Float16*)ws;           ws += (size_t)BDIM * SDIM * CEMB * 2;
  _Float16* feat16 = (_Float16*)ws;           ws += (size_t)BDIM * SDIM * FEATD * 2;
  float*    Z      = (float*)ws;              ws += (size_t)BDIM * SDIM * NZ * 4;
  _Float16* Wc16   = (_Float16*)ws;           ws += (size_t)CEMB * NZ * 2;
  float*    biasC  = (float*)ws;              ws += NZ * 4;
  _Float16* U16f   = (_Float16*)ws;           ws += (size_t)HID * GATES * 2;
  _Float16* U16b   = (_Float16*)ws;           ws += (size_t)HID * GATES * 2;
  _Float16* WgAll  = (_Float16*)ws;           ws += (size_t)FEATD * NBA * 2;
  float*    bgAll  = (float*)ws;              ws += NBA * 4;
  _Float16* ev16   = (_Float16*)ws;           ws += (size_t)BDIM * EV_EMBED * 2;

  // 1) embeddings + mask
  k_embed<<<(BDIM * SDIM * CEMB) / 256, 256, 0, stream>>>(tokens, table, emb16, out_mask);
  // 2) weight conversions
  k_prep_weights<<<64, 256, 0, stream>>>(Wf, bf, Wb, bb, Uf, Ub, Wc16, biasC, U16f, U16b);
  k_prep_expert<<<64, 256, 0, stream>>>(ev_type, W_arg, b_arg, ev_table, WgAll, bgAll, ev16);
  // 3) event embedding broadcast into feat tail
  k_fill_ev<<<(BDIM * SDIM * EV_EMBED) / 256, 256, 0, stream>>>(ev16, feat16);
  // 4) input projections (WMMA GEMM)
  k_xw_gemm<<<dim3((BDIM * SDIM) / 64, NZ / 64), 256, 0, stream>>>(emb16, Wc16, biasC, Z);
  // 5) the two recurrences (1 block each, WMMA per step, pipelined Z staging)
  const size_t lstm_smem = 36864 + 9216 + 16384 + 131072 + 512;   // 194,048 B
  k_lstm<<<2, 1024, lstm_smem, stream>>>(Z, U16f, U16b, tokens, feat16);
  // 6) event logits from last timestep
  k_event_logits<<<1, 1024, 0, stream>>>(feat16, Wt, bt, out_logits);
  // 7) big expert einsum (WMMA GEMM)
  k_arg_gemm<<<dim3((BDIM * SDIM) / 64, NBA / 64), 256, 0, stream>>>(feat16, WgAll, bgAll, out_args);
}
